// E2ECR_60224031424680
// MI455X (gfx1250) — compile-verified, matrix operand order checked by reading, not run on hardware
//
#include <hip/hip_runtime.h>
#include <hip/hip_bf16.h>
#include <stdint.h>

// ---------------- problem constants ----------------
#define Bb   2
#define Tt   2048
#define Ee   256
#define Ss   128
#define Ww   8
#define Hh   150     // MLP hidden width
#define HP   160     // padded to 10 x 16 WMMA tiles
#define K3E  768     // 3*E

typedef __bf16 bf16_t;
typedef __attribute__((ext_vector_type(16))) __bf16 bf16x16;
typedef __attribute__((ext_vector_type(8)))  float  floatx8;

__device__ __forceinline__ bf16_t f2b(float x) { return (bf16_t)x; }
__device__ __forceinline__ float  b2f(bf16_t x){ return (float)x; }

__device__ __forceinline__ floatx8 zero8() { floatx8 z = {}; return z; }

union Frag {
    bf16x16 v;
    uint4   q[2];
};

union Half8 {
    uint4  q;
    bf16_t e[8];
};

// Load a 16x32 bf16 fragment (A layout; B uses the mirrored convention with
// lane = N and the matrix stored N x K row-major).
// Lanes 0-15 : row m=lane,    K k0+0..7  and k0+16..23
// Lanes 16-31: row m=lane-16, K k0+8..15 and k0+24..31
template <typename P>
__device__ __forceinline__ Frag load_frag(const P* base, int r0, int k0, int ld, int lane) {
    Frag f;
    const int m  = lane & 15;
    const int ks = lane >> 4;          // 0 or 1
    const P* p = base + (size_t)(r0 + m) * ld + k0 + ks * 8;
    f.q[0] = *(const uint4*)(p);       // K +0..7   (16B)
    f.q[1] = *(const uint4*)(p + 16);  // K +16..23 (16B)
    return f;
}

__device__ __forceinline__ floatx8 wmma_bf16(const Frag& a, const Frag& b, floatx8 c) {
    return __builtin_amdgcn_wmma_f32_16x16x32_bf16(
        /*neg_a=*/false, a.v, /*neg_b=*/false, b.v,
        /*c_mod=*/(short)0, c, /*reuse_a=*/false, /*reuse_b=*/false);
}

// ---------------- prep kernels ----------------

__global__ void k_f2b(const float* __restrict__ x, bf16_t* __restrict__ y, int n) {
    int i = blockIdx.x * blockDim.x + threadIdx.x;
    if (i < n) y[i] = f2b(x[i]);
}

// w: (K x Hh) row-major fp32 -> out: (HP x K) row-major bf16, zero padded.
// Ksrc: valid K range in source (for HP-padded K dims).
__global__ void k_transpose_pad(const float* __restrict__ w, bf16_t* __restrict__ out,
                                int K, int Ksrc) {
    int idx = blockIdx.x * blockDim.x + threadIdx.x;
    if (idx >= HP * K) return;
    int n = idx / K, k = idx - n * K;
    float v = (n < Hh && k < Ksrc) ? w[(size_t)k * Hh + n] : 0.f;
    out[idx] = f2b(v);
}

__global__ void k_pad_bias(const float* __restrict__ b, float* __restrict__ out) {
    int i = threadIdx.x;
    out[i] = (i < Hh) ? b[i] : 0.f;
}

// ---------------- generic 2-hidden-layer MLP -> scalar, WMMA ----------------
// X: rows x KIN bf16, w1T: HP x KIN, w2T: HP x HP, out: rows fp32.
// 128 threads = 4 waves, each wave owns one 16-row tile (64 rows / block).
template <int KIN>
__global__ __launch_bounds__(128) void k_mlp_scalar(
    const bf16_t* __restrict__ X,
    const bf16_t* __restrict__ w1T, const float* __restrict__ b1p,
    const bf16_t* __restrict__ w2T, const float* __restrict__ b2p,
    const float* __restrict__ w3, const float* __restrict__ b3,
    float* __restrict__ out) {
    __shared__ __align__(16) bf16_t Hs[4][16 * HP];

    const int lane   = threadIdx.x & 31;
    const int wv     = threadIdx.x >> 5;
    const int row0   = (blockIdx.x * 4 + wv) * 16;
    const int col    = lane & 15;
    const int rowoff = (lane >> 4) * 8;

    floatx8 acc[10];
#pragma unroll
    for (int t = 0; t < 10; ++t) acc[t] = zero8();

    // layer 1: stage all 10 B-fragments, then run the WMMA chain
    for (int k0 = 0; k0 < KIN; k0 += 32) {
        Frag a = load_frag(X, row0, k0, KIN, lane);
        Frag bf[10];
#pragma unroll
        for (int t = 0; t < 10; ++t) bf[t] = load_frag(w1T, t * 16, k0, KIN, lane);
#pragma unroll
        for (int t = 0; t < 10; ++t) acc[t] = wmma_bf16(a, bf[t], acc[t]);
    }
    bf16_t* hs = &Hs[wv][0];
#pragma unroll
    for (int t = 0; t < 10; ++t) {
        int   n  = t * 16 + col;
        float bb = b1p[n];
#pragma unroll
        for (int r = 0; r < 8; ++r) {
            float v = acc[t][r] + bb;
            hs[(size_t)(rowoff + r) * HP + n] = f2b(v > 0.f ? v : 0.f);
        }
        acc[t] = zero8();
    }
    // layer 2 (A from per-wave LDS staging)
    for (int k0 = 0; k0 < HP; k0 += 32) {
        Frag a = load_frag(hs, 0, k0, HP, lane);
        Frag bf[10];
#pragma unroll
        for (int t = 0; t < 10; ++t) bf[t] = load_frag(w2T, t * 16, k0, HP, lane);
#pragma unroll
        for (int t = 0; t < 10; ++t) acc[t] = wmma_bf16(a, bf[t], acc[t]);
    }
#pragma unroll
    for (int t = 0; t < 10; ++t) {
        int   n  = t * 16 + col;
        float bb = b2p[n];
#pragma unroll
        for (int r = 0; r < 8; ++r) {
            float v = acc[t][r] + bb;
            hs[(size_t)(rowoff + r) * HP + n] = f2b(v > 0.f ? v : 0.f);
        }
    }
    // layer 3: scalar head
    if (lane < 16) {
        float s = b3[0];
        for (int h = 0; h < Hh; ++h) s += b2f(hs[(size_t)lane * HP + h]) * w3[h];
        out[row0 + lane] = s;
    }
}

// ---------------- span gather: build g_i (start|end|weighted) ----------------
__global__ void k_gather(const float* __restrict__ emb, const int* __restrict__ starts,
                         const float* __restrict__ attns,
                         float* __restrict__ g, bf16_t* __restrict__ g16) {
    const int bs = blockIdx.x;           // b*S + s
    const int b  = bs / Ss;
    const int e  = threadIdx.x;          // 0..E-1
    const int tok0 = starts[bs];
    const float* eb = emb + (size_t)b * Tt * Ee;
    const float* at = attns + (size_t)b * Tt;

    float st = eb[(size_t)tok0 * Ee + e];
    float en = eb[(size_t)(tok0 + Ww - 1) * Ee + e];
    float ws = 0.f;
#pragma unroll
    for (int w = 0; w < Ww; ++w) ws += eb[(size_t)(tok0 + w) * Ee + e] * at[tok0 + w];

    float*  go   = g   + (size_t)bs * K3E;
    bf16_t* go16 = g16 + (size_t)bs * K3E;
    go[e] = st;            go[Ee + e] = en;            go[2 * Ee + e] = ws;
    go16[e] = f2b(st);     go16[Ee + e] = f2b(en);     go16[2 * Ee + e] = f2b(ws);
}

// ---------------- Aib = gi @ pair_w1[0:768] + pair_b1 (fp32, tiny) ----------
__global__ void k_aib(const float* __restrict__ g, const float* __restrict__ pw1,
                      const float* __restrict__ pb1, float* __restrict__ aib) {
    const int row = blockIdx.x;          // b*S + i
    const int h   = threadIdx.x;         // 0..HP-1
    float s = 0.f;
    if (h < Hh) {
        s = pb1[h];
        const float* gr = g + (size_t)row * K3E;
        for (int k = 0; k < K3E; ++k) s += gr[k] * pw1[(size_t)k * Hh + h];
    }
    aib[(size_t)row * HP + h] = s;
}

// ---------------- pair kernel: one block per (b,i) ----------------
// h1[j,:] = relu( G[b] @ (gi*w1c + w1b)  +  Aib[i,:] )
// h2 = relu(h1 @ w2 + b2);  score = h2 @ w3 + b3
// out[b,i,j] = (ment[i] + ment[j] + score)/3
__global__ __launch_bounds__(256) void k_pair(
    const bf16_t* __restrict__ G16,    // (B*S) x 768 bf16
    const float*  __restrict__ g,      // (B*S) x 768 fp32 (for gi)
    const bf16_t* __restrict__ w1bT,   // HP x 768
    const bf16_t* __restrict__ w1cT,   // HP x 768
    const float*  __restrict__ aib,    // (B*S) x HP (includes b1)
    const bf16_t* __restrict__ w2T,    // HP x HP
    const float*  __restrict__ b2p,    // HP
    const float*  __restrict__ w3, const float* __restrict__ b3,
    const float*  __restrict__ ment,   // B*S
    float* __restrict__ out) {         // (B*S) x S
    __shared__ __align__(16) bf16_t Bs[HP][64];      // 20 KB k-panel of Bmat^T
    __shared__ __align__(16) bf16_t Hs[8][16 * HP];  // 40 KB per-wave staging
    __shared__ __align__(16) float gis[K3E];

    const int bi   = blockIdx.x;        // b*S + i
    const int b    = bi / Ss;
    const int tid  = threadIdx.x;
    const int lane = tid & 31;
    const int wv   = tid >> 5;
    const int col    = lane & 15;
    const int rowoff = (lane >> 4) * 8;
    const int j0   = wv * 16;

    for (int k = tid; k < K3E; k += 256) gis[k] = g[(size_t)bi * K3E + k];
    __syncthreads();

    floatx8 acc[10];
#pragma unroll
    for (int t = 0; t < 10; ++t) acc[t] = zero8();

    const bf16_t* Gb = G16 + (size_t)b * Ss * K3E;

    for (int kb = 0; kb < K3E; kb += 64) {
        // build Bmat panel, 8 elems per op:
        // Bs[h][k] = gi[kb+k]*w1c[kb+k,h] + w1b[kb+k,h]
        for (int idx = tid; idx < HP * 8; idx += 256) {
            const int h  = idx >> 3;
            const int kg = (idx & 7) * 8;
            Half8 c8, b8, o8;
            c8.q = *(const uint4*)&w1cT[(size_t)h * K3E + kb + kg];
            b8.q = *(const uint4*)&w1bT[(size_t)h * K3E + kb + kg];
            const float4 g0 = *(const float4*)&gis[kb + kg];
            const float4 g1 = *(const float4*)&gis[kb + kg + 4];
            const float gg[8] = {g0.x, g0.y, g0.z, g0.w, g1.x, g1.y, g1.z, g1.w};
#pragma unroll
            for (int u = 0; u < 8; ++u)
                o8.e[u] = f2b(gg[u] * b2f(c8.e[u]) + b2f(b8.e[u]));
            *(uint4*)&Bs[h][kg] = o8.q;
        }
        __syncthreads();
#pragma unroll
        for (int ks = 0; ks < 64; ks += 32) {
            Frag a = load_frag(Gb, j0, kb + ks, K3E, lane);
            Frag bf[10];
#pragma unroll
            for (int t = 0; t < 10; ++t) bf[t] = load_frag(&Bs[0][0], t * 16, ks, 64, lane);
#pragma unroll
            for (int t = 0; t < 10; ++t) acc[t] = wmma_bf16(a, bf[t], acc[t]);
        }
        __syncthreads();
    }

    // + Aib (includes b1), relu -> Hs (bf16, A-layout staging)
    const float* ai = aib + (size_t)bi * HP;
    bf16_t* hs = &Hs[wv][0];
#pragma unroll
    for (int t = 0; t < 10; ++t) {
        int   n  = t * 16 + col;
        float ab = ai[n];
#pragma unroll
        for (int r = 0; r < 8; ++r) {
            float v = acc[t][r] + ab;
            hs[(size_t)(rowoff + r) * HP + n] = f2b(v > 0.f ? v : 0.f);
        }
        acc[t] = zero8();
    }
    // layer 2
    for (int k0 = 0; k0 < HP; k0 += 32) {
        Frag a = load_frag(hs, 0, k0, HP, lane);
        Frag bf[10];
#pragma unroll
        for (int t = 0; t < 10; ++t) bf[t] = load_frag(w2T, t * 16, k0, HP, lane);
#pragma unroll
        for (int t = 0; t < 10; ++t) acc[t] = wmma_bf16(a, bf[t], acc[t]);
    }
#pragma unroll
    for (int t = 0; t < 10; ++t) {
        int   n  = t * 16 + col;
        float bb = b2p[n];
#pragma unroll
        for (int r = 0; r < 8; ++r) {
            float v = acc[t][r] + bb;
            hs[(size_t)(rowoff + r) * HP + n] = f2b(v > 0.f ? v : 0.f);
        }
    }
    // layer 3 + combine with mention scores
    if (lane < 16) {
        const int j = j0 + lane;
        float s = b3[0];
        for (int h = 0; h < Hh; ++h) s += b2f(hs[(size_t)lane * HP + h]) * w3[h];
        float mi = ment[bi];
        float mj = ment[(size_t)b * Ss + j];
        out[(size_t)bi * Ss + j] = (mi + mj + s) * (1.0f / 3.0f);
    }
}

// ---------------- host side ----------------
extern "C" void kernel_launch(void* const* d_in, const int* in_sizes, int n_in,
                              void* d_out, int out_size, void* d_ws, size_t ws_size,
                              hipStream_t stream) {
    (void)in_sizes; (void)n_in; (void)out_size; (void)ws_size;
    const float* emb    = (const float*)d_in[0];
    const int*   starts = (const int*)d_in[1];
    const float* aw1 = (const float*)d_in[2];  const float* ab1 = (const float*)d_in[3];
    const float* aw2 = (const float*)d_in[4];  const float* ab2 = (const float*)d_in[5];
    const float* aw3 = (const float*)d_in[6];  const float* ab3 = (const float*)d_in[7];
    const float* mw1 = (const float*)d_in[8];  const float* mb1 = (const float*)d_in[9];
    const float* mw2 = (const float*)d_in[10]; const float* mb2 = (const float*)d_in[11];
    const float* mw3 = (const float*)d_in[12]; const float* mb3 = (const float*)d_in[13];
    const float* pw1 = (const float*)d_in[14]; const float* pb1 = (const float*)d_in[15];
    const float* pw2 = (const float*)d_in[16]; const float* pb2 = (const float*)d_in[17];
    const float* pw3 = (const float*)d_in[18]; const float* pb3 = (const float*)d_in[19];
    float* out = (float*)d_out;

    char* p = (char*)d_ws;
    auto carve = [&](size_t bytes) -> char* {
        char* r = p;
        p += (bytes + 255) & ~(size_t)255;
        return r;
    };
    bf16_t* X16  = (bf16_t*)carve((size_t)Bb * Tt * Ee * 2);
    bf16_t* G16  = (bf16_t*)carve((size_t)Bb * Ss * K3E * 2);
    float*  gbuf = (float*) carve((size_t)Bb * Ss * K3E * 4);
    float*  attn = (float*) carve((size_t)Bb * Tt * 4);
    float*  mnt  = (float*) carve((size_t)Bb * Ss * 4);
    float*  Aib  = (float*) carve((size_t)Bb * Ss * HP * 4);
    bf16_t* aw1T = (bf16_t*)carve((size_t)HP * Ee * 2);
    bf16_t* aw2T = (bf16_t*)carve((size_t)HP * HP * 2);
    bf16_t* mw1T = (bf16_t*)carve((size_t)HP * K3E * 2);
    bf16_t* mw2T = (bf16_t*)carve((size_t)HP * HP * 2);
    bf16_t* w1bT = (bf16_t*)carve((size_t)HP * K3E * 2);
    bf16_t* w1cT = (bf16_t*)carve((size_t)HP * K3E * 2);
    bf16_t* pw2T = (bf16_t*)carve((size_t)HP * HP * 2);
    float* ab1p = (float*)carve(HP * 4);
    float* ab2p = (float*)carve(HP * 4);
    float* mb1p = (float*)carve(HP * 4);
    float* mb2p = (float*)carve(HP * 4);
    float* pb2p = (float*)carve(HP * 4);

    // ---- prep: bf16 conversions, transposed/padded weights, padded biases
    {
        int n = Bb * Tt * Ee;
        k_f2b<<<(n + 255) / 256, 256, 0, stream>>>(emb, X16, n);
    }
    { int t = HP * Ee;  k_transpose_pad<<<(t + 255) / 256, 256, 0, stream>>>(aw1, aw1T, Ee, Ee); }
    { int t = HP * HP;  k_transpose_pad<<<(t + 255) / 256, 256, 0, stream>>>(aw2, aw2T, HP, Hh); }
    { int t = HP * K3E; k_transpose_pad<<<(t + 255) / 256, 256, 0, stream>>>(mw1, mw1T, K3E, K3E); }
    { int t = HP * HP;  k_transpose_pad<<<(t + 255) / 256, 256, 0, stream>>>(mw2, mw2T, HP, Hh); }
    { int t = HP * K3E; k_transpose_pad<<<(t + 255) / 256, 256, 0, stream>>>(pw1 + (size_t)1 * K3E * Hh, w1bT, K3E, K3E); }
    { int t = HP * K3E; k_transpose_pad<<<(t + 255) / 256, 256, 0, stream>>>(pw1 + (size_t)2 * K3E * Hh, w1cT, K3E, K3E); }
    { int t = HP * HP;  k_transpose_pad<<<(t + 255) / 256, 256, 0, stream>>>(pw2, pw2T, HP, Hh); }
    k_pad_bias<<<1, HP, 0, stream>>>(ab1, ab1p);
    k_pad_bias<<<1, HP, 0, stream>>>(ab2, ab2p);
    k_pad_bias<<<1, HP, 0, stream>>>(mb1, mb1p);
    k_pad_bias<<<1, HP, 0, stream>>>(mb2, mb2p);
    k_pad_bias<<<1, HP, 0, stream>>>(pb2, pb2p);

    // ---- attn MLP over all B*T tokens (WMMA)
    k_mlp_scalar<Ee><<<(Bb * Tt) / 64, 128, 0, stream>>>(X16, aw1T, ab1p, aw2T, ab2p,
                                                         aw3, ab3, attn);
    // ---- span gather -> g_i
    k_gather<<<Bb * Ss, Ee, 0, stream>>>(emb, starts, attn, gbuf, G16);

    // ---- mention MLP (WMMA)
    k_mlp_scalar<K3E><<<(Bb * Ss) / 64, 128, 0, stream>>>(G16, mw1T, mb1p, mw2T, mb2p,
                                                          mw3, mb3, mnt);
    // ---- Aib = gi @ w1a + b1 (tiny, fp32)
    k_aib<<<Bb * Ss, HP, 0, stream>>>(gbuf, pw1, pb1, Aib);

    // ---- pair MLP, factored: one block per (b,i)
    k_pair<<<Bb * Ss, 256, 0, stream>>>(G16, gbuf, w1bT, w1cT, Aib, pw2T, pb2p,
                                        pw3, pb3, mnt, out);
}
